// TemporalRelationGraph_81097572483747
// MI455X (gfx1250) — compile-verified, matrix operand order checked by reading, not run on hardware
//
#include <hip/hip_runtime.h>
#include <vector>
#include <array>
#include <algorithm>
#include <cmath>
#include <cstdint>
#include <cstddef>

// ---------------------------------------------------------------------------
// Types for CDNA5 WMMA (wave32): 16x16x32 bf16 -> f32
// ---------------------------------------------------------------------------
typedef __attribute__((ext_vector_type(8)))  __bf16 v8bf;
typedef __attribute__((ext_vector_type(16))) __bf16 v16bf;
typedef __attribute__((ext_vector_type(8)))  float  v8f;

#define DEVINL __device__ __forceinline__

struct SelArg { int idx[3][4]; };   // relation table: up to 3 rows x 4 frame ids

DEVINL unsigned short f32_to_bf16_bits(float f) {
  unsigned int u = __float_as_uint(f);
  u += 0x7fffu + ((u >> 16) & 1u);          // round-to-nearest-even
  return (unsigned short)(u >> 16);
}

// 16-bit A/B fragment layout (ISA 7.12.2): lane (m|n = lane%16, half = lane/16)
// holds K = {8*half + 0..7, 16 + 8*half + 0..7}  -> two contiguous 16B chunks.
DEVINL v16bf load_frag(const unsigned short* __restrict__ p, int off, int half) {
  v8bf lo = *reinterpret_cast<const v8bf*>(p + off + 8 * half);
  v8bf hi = *reinterpret_cast<const v8bf*>(p + off + 16 + 8 * half);
  return __builtin_shufflevector(lo, hi, 0,1,2,3,4,5,6,7,8,9,10,11,12,13,14,15);
}

DEVINL v8f wmma_bf16(v16bf a, v16bf b, v8f c) {
  return __builtin_amdgcn_wmma_f32_16x16x32_bf16(false, a, false, b,
                                                 (short)0, c, false, false);
}

// ---------------------------------------------------------------------------
// Stage 0a: Xr[b,t,:] = bf16(relu(concat(batch_variation, graph_out)))
// ---------------------------------------------------------------------------
__global__ __launch_bounds__(256)
void prep_xr_kernel(const float* __restrict__ graph_out,
                    const float* __restrict__ batch_var,
                    unsigned short* __restrict__ Xr) {
  const int bt = blockIdx.x;  // b*16 + t, 0..16383
  for (int d = threadIdx.x; d < 1280; d += blockDim.x) {
    float v = (d < 1024) ? batch_var[(size_t)bt * 1024 + d]
                         : graph_out[(size_t)bt * 256 + (d - 1024)];
    v = v > 0.f ? v : 0.f;
    Xr[(size_t)bt * 1280 + d] = f32_to_bf16_bits(v);
  }
}

// ---------------------------------------------------------------------------
// Stage 0b: Wt[n, k] = bf16(W[k, n]), zero-padded rows for n >= N.
// ---------------------------------------------------------------------------
__global__ __launch_bounds__(256)
void convert_wt_kernel(const float* __restrict__ src,
                       unsigned short* __restrict__ dst,
                       int K, int N) {
  const int n = blockIdx.x;   // 0 .. Npad-1
  for (int k = threadIdx.x; k < K; k += blockDim.x) {
    float v = (n < N) ? src[(size_t)k * N + n] : 0.f;
    dst[(size_t)n * K + k] = f32_to_bf16_bits(v);
  }
}

// ---------------------------------------------------------------------------
// Layers 1 & 2: Out(M,256) = bf16(relu(A @ W^T + bias)).
// GATHER: A row (r*1024+b), k = j*1280+d  reads Xr[b, sel[r][j], d].
// Block = 128 threads = 4 waves; wave computes a 32(M) x 64(N) macro-tile
// (2 m-subtiles x 4 n-subtiles => 12 b128 loads per 8 WMMAs).
// ---------------------------------------------------------------------------
template<bool GATHER>
__global__ __launch_bounds__(128)
void gemm12_kernel(const unsigned short* __restrict__ A,
                   const unsigned short* __restrict__ Bt,
                   const float* __restrict__ bias,
                   unsigned short* __restrict__ Out,
                   int K, int s, SelArg sel) {
  const int lane = threadIdx.x & 31;
  const int wave = threadIdx.x >> 5;
  const int half = lane >> 4;
  const int l16  = lane & 15;
  const int m0 = blockIdx.x * 128 + wave * 32;   // two 16-row subtiles m0, m0+16
  const int n0 = blockIdx.y * 64;

  const unsigned short* brow[4];
#pragma unroll
  for (int t = 0; t < 4; ++t)
    brow[t] = Bt + (size_t)(n0 + t * 16 + l16) * K;

  v8f acc0[4] = {};
  v8f acc1[4] = {};

  if (GATHER) {
    const int r = m0 >> 10;                 // uniform: 128-tiles never cross B=1024
    const int b = (m0 & 1023) + l16;
    for (int j = 0; j < s; ++j) {
      const int frame = sel.idx[r][j];
      const unsigned short* arow0 = A + (size_t)(((b)      << 4) + frame) * 1280;
      const unsigned short* arow1 = A + (size_t)(((b + 16) << 4) + frame) * 1280;
      const int kb = j * 1280;
#pragma unroll 1
      for (int d = 0; d < 1280; d += 32) {
        v16bf af0 = load_frag(arow0, d, half);
        v16bf af1 = load_frag(arow1, d, half);
#pragma unroll
        for (int t = 0; t < 4; ++t) {
          v16bf bf = load_frag(brow[t], kb + d, half);
          acc0[t] = wmma_bf16(af0, bf, acc0[t]);
          acc1[t] = wmma_bf16(af1, bf, acc1[t]);
        }
      }
    }
  } else {
    const unsigned short* arow0 = A + (size_t)(m0 + l16) * K;
    const unsigned short* arow1 = A + (size_t)(m0 + 16 + l16) * K;
#pragma unroll 1
    for (int k = 0; k < K; k += 32) {
      v16bf af0 = load_frag(arow0, k, half);
      v16bf af1 = load_frag(arow1, k, half);
#pragma unroll
      for (int t = 0; t < 4; ++t) {
        v16bf bf = load_frag(brow[t], k, half);
        acc0[t] = wmma_bf16(af0, bf, acc0[t]);
        acc1[t] = wmma_bf16(af1, bf, acc1[t]);
      }
    }
  }

  // C/D layout: lane -> n = l16, element v -> m = v + 8*half
#pragma unroll
  for (int t = 0; t < 4; ++t) {
    const int n = n0 + t * 16 + l16;
    const float bn = bias[n];
#pragma unroll
    for (int v = 0; v < 8; ++v) {
      const int m = m0 + v + half * 8;
      float val0 = acc0[t][v] + bn;
      float val1 = acc1[t][v] + bn;
      val0 = val0 > 0.f ? val0 : 0.f;
      val1 = val1 > 0.f ? val1 : 0.f;
      Out[(size_t)m * 256 + n]        = f32_to_bf16_bits(val0);
      Out[(size_t)(m + 16) * 256 + n] = f32_to_bf16_bits(val1);
    }
  }
}

// ---------------------------------------------------------------------------
// Layer 3 + reduce over relation rows:
//   out[b*8 + ratio, n] (+)= sum_r H2[r*1024+b,:] @ W3 + n_sel*b3[n]
// k outer / r inner: W3 fragments loaded once per k-step, reused 2*n_sel times.
// W3t padded to N=448 (grid.y = 7 tiles of 64); store guard n < 400.
// ---------------------------------------------------------------------------
__global__ __launch_bounds__(128)
void gemm3_kernel(const unsigned short* __restrict__ H2,
                  const unsigned short* __restrict__ W3t,
                  const float* __restrict__ b3,
                  float* __restrict__ out,
                  int n_sel, int ratio_m1, int accum) {
  const int lane = threadIdx.x & 31;
  const int wave = threadIdx.x >> 5;
  const int half = lane >> 4;
  const int l16  = lane & 15;
  const int b0 = blockIdx.x * 128 + wave * 32;   // two 16-row subtiles
  const int n0 = blockIdx.y * 64;

  const unsigned short* brow[4];
#pragma unroll
  for (int t = 0; t < 4; ++t)
    brow[t] = W3t + (size_t)(n0 + t * 16 + l16) * 256;

  const unsigned short* arow0 = H2 + (size_t)(b0 + l16) * 256;
  const unsigned short* arow1 = H2 + (size_t)(b0 + 16 + l16) * 256;

  v8f acc0[4] = {};
  v8f acc1[4] = {};

#pragma unroll 1
  for (int k = 0; k < 256; k += 32) {
    v16bf bf0 = load_frag(brow[0], k, half);
    v16bf bf1 = load_frag(brow[1], k, half);
    v16bf bf2 = load_frag(brow[2], k, half);
    v16bf bf3 = load_frag(brow[3], k, half);
#pragma unroll
    for (int r = 0; r < 3; ++r) {
      if (r >= n_sel) break;                // uniform branch
      const size_t roff = (size_t)(r << 10) * 256;
      v16bf af0 = load_frag(arow0 + roff, k, half);
      v16bf af1 = load_frag(arow1 + roff, k, half);
      acc0[0] = wmma_bf16(af0, bf0, acc0[0]);
      acc0[1] = wmma_bf16(af0, bf1, acc0[1]);
      acc0[2] = wmma_bf16(af0, bf2, acc0[2]);
      acc0[3] = wmma_bf16(af0, bf3, acc0[3]);
      acc1[0] = wmma_bf16(af1, bf0, acc1[0]);
      acc1[1] = wmma_bf16(af1, bf1, acc1[1]);
      acc1[2] = wmma_bf16(af1, bf2, acc1[2]);
      acc1[3] = wmma_bf16(af1, bf3, acc1[3]);
    }
  }

#pragma unroll
  for (int t = 0; t < 4; ++t) {
    const int n = n0 + t * 16 + l16;
    if (n < 400) {                          // whole 16-wide subtile valid/invalid
      const float bn = b3[n] * (float)n_sel;
#pragma unroll
      for (int v = 0; v < 8; ++v) {
        const int b = b0 + v + half * 8;
        const size_t o0 = ((size_t)b * 8 + ratio_m1) * 400 + n;
        const size_t o1 = ((size_t)(b + 16) * 8 + ratio_m1) * 400 + n;
        float val0 = acc0[t][v] + bn;
        float val1 = acc1[t][v] + bn;
        out[o0] = accum ? (out[o0] + val0) : val0;
        out[o1] = accum ? (out[o1] + val1) : val1;
      }
    }
  }
}

// ---------------------------------------------------------------------------
// Host: reproduce SEL (itertools.combinations lex order + np.round half-even)
// ---------------------------------------------------------------------------
struct Sel { int n; int idx[3][4]; };

static void build_sel(Sel sel[8][4]) {
  static const int scales[4] = {4, 3, 2, 1};
  for (int ratio = 1; ratio <= 8; ++ratio) {
    const int seq = 2 * ratio;
    for (int si = 0; si < 4; ++si) {
      const int s = scales[si];
      Sel& S = sel[ratio - 1][si];
      S.n = 0;
      if (s > seq) continue;
      std::vector<std::array<int,4>> combos;
      int c[4];
      for (int i = 0; i < s; ++i) c[i] = i;
      while (true) {
        std::array<int,4> a{0,0,0,0};
        for (int i = 0; i < s; ++i) a[i] = c[i];
        combos.push_back(a);
        int i = s - 1;
        while (i >= 0 && c[i] == seq - s + i) --i;
        if (i < 0) break;
        ++c[i];
        for (int j = i + 1; j < s; ++j) c[j] = c[j - 1] + 1;
      }
      const int n_rel = (int)combos.size();
      const int k = std::min(3, n_rel);
      std::vector<int> ids;
      for (int i = 0; i < k; ++i) {
        double v = (k == 1) ? 0.0 : (double)(n_rel - 1) * i / (double)(k - 1);
        ids.push_back((int)std::nearbyint(v));   // RNE, matches np.round
      }
      std::sort(ids.begin(), ids.end());
      ids.erase(std::unique(ids.begin(), ids.end()), ids.end());
      S.n = (int)ids.size();
      for (int i = 0; i < S.n; ++i)
        for (int j = 0; j < s; ++j) S.idx[i][j] = combos[ids[i]][j];
    }
  }
}

// ---------------------------------------------------------------------------
// kernel_launch
// d_in: [0]=graph_out (B,16,256) f32, [1]=batch_variation (B,16,1024) f32,
//       then per scale si: W1,b1,W2,b2,W3,b3 at indices 2+6*si ..
// ---------------------------------------------------------------------------
extern "C" void kernel_launch(void* const* d_in, const int* in_sizes, int n_in,
                              void* d_out, int out_size, void* d_ws, size_t ws_size,
                              hipStream_t stream) {
  (void)in_sizes; (void)n_in; (void)out_size; (void)ws_size;
  static const int scales[4] = {4, 3, 2, 1};

  const float* graph_out = (const float*)d_in[0];
  const float* batch_var = (const float*)d_in[1];

  // Carve workspace (256B aligned). Total ~53 MB.
  uintptr_t base = (uintptr_t)d_ws;
  auto alloc_us = [&](size_t elems) {
    uintptr_t p = base;
    base += ((elems * 2 + 255) & ~(size_t)255);
    return (unsigned short*)p;
  };
  unsigned short* Xr = alloc_us((size_t)16384 * 1280);          // relu'd input, bf16
  unsigned short *W1t[4], *W2t[4], *W3t[4];
  for (int si = 0; si < 4; ++si) {
    W1t[si] = alloc_us((size_t)256 * scales[si] * 1280);
    W2t[si] = alloc_us((size_t)256 * 256);
    W3t[si] = alloc_us((size_t)448 * 256);                      // N padded to 448
  }
  unsigned short* H1 = alloc_us((size_t)3072 * 256);
  unsigned short* H2 = alloc_us((size_t)3072 * 256);

  prep_xr_kernel<<<16384, 256, 0, stream>>>(graph_out, batch_var, Xr);
  for (int si = 0; si < 4; ++si) {
    const float* W1 = (const float*)d_in[2 + si * 6 + 0];
    const float* W2 = (const float*)d_in[2 + si * 6 + 2];
    const float* W3 = (const float*)d_in[2 + si * 6 + 4];
    convert_wt_kernel<<<256, 256, 0, stream>>>(W1, W1t[si], scales[si] * 1280, 256);
    convert_wt_kernel<<<256, 256, 0, stream>>>(W2, W2t[si], 256, 256);
    convert_wt_kernel<<<448, 128, 0, stream>>>(W3, W3t[si], 256, 400);
  }

  Sel sel[8][4];
  build_sel(sel);

  for (int ratio = 1; ratio <= 8; ++ratio) {
    bool first = true;
    for (int si = 0; si < 4; ++si) {
      const int s = scales[si];
      if (s > 2 * ratio) continue;          // adjust_scale: dropped scales
      const Sel& S = sel[ratio - 1][si];
      const int n_sel = S.n;
      const int M = n_sel * 1024;

      SelArg sa{};
      for (int r = 0; r < n_sel; ++r)
        for (int j = 0; j < s; ++j) sa.idx[r][j] = S.idx[r][j];

      const float* b1 = (const float*)d_in[2 + si * 6 + 1];
      const float* b2 = (const float*)d_in[2 + si * 6 + 3];
      const float* b3 = (const float*)d_in[2 + si * 6 + 5];

      gemm12_kernel<true ><<<dim3(M / 128, 4), 128, 0, stream>>>(
          Xr, W1t[si], b1, H1, s * 1280, s, sa);
      SelArg z{};
      gemm12_kernel<false><<<dim3(M / 128, 4), 128, 0, stream>>>(
          H1, W2t[si], b2, H2, 256, 0, z);
      gemm3_kernel<<<dim3(8, 7), 128, 0, stream>>>(
          H2, W3t[si], b3, (float*)d_out, n_sel, ratio - 1, first ? 0 : 1);
      first = false;
    }
  }
}
